// ToyModel_9869834846219
// MI455X (gfx1250) — compile-verified, matrix operand order checked by reading, not run on hardware
//
#include <hip/hip_runtime.h>
#include <hip/hip_bf16.h>

#define B_    8192
#define F_    256
#define HALF_ 128
#define H_    512
#define L_    4
#define K_    32768
#define KSLICES 8
#define CODES_PER_SLICE (K_/KSLICES)      // 4096
#define NT_PER_SLICE (CODES_PER_SLICE/16) // 256

typedef __attribute__((ext_vector_type(16))) __bf16 v16bf;
typedef __attribute__((ext_vector_type(8)))  float  v8f;
typedef unsigned int v4u __attribute__((ext_vector_type(4)));

union FragU { v16bf v; v4u q[2]; };

// Load one 16x32 bf16 A/B fragment slice for this lane: two contiguous
// 16-byte chunks (K-contiguous memory), per the CDNA5 16-bit WMMA layout
// (lanes 0-15 hold K chunks {0..7,16..23}, lanes 16-31 hold {8..15,24..31}).
__device__ inline v16bf ldfrag(const __bf16* p0, const __bf16* p1) {
  FragU f;
  f.q[0] = *(const v4u*)p0;
  f.q[1] = *(const v4u*)p1;
  return f.v;
}

// ---- CDNA5 async global->LDS copy (ASYNCcnt path), with safe fallback ----
#if __has_builtin(__builtin_amdgcn_global_load_async_to_lds_b128)
#define ASYNC_LDS 1
typedef int v4i __attribute__((vector_size(16)));
typedef v4i __attribute__((address_space(1))) as1_v4i;  // global (prints as __device__)
typedef v4i __attribute__((address_space(3))) as3_v4i;  // LDS
__device__ inline void async_cp16(const void* g, void* l) {
  // generic-LDS low 32 bits == LDS offset (ISA aperture rule); global generic == AS1.
  as1_v4i* gp = (as1_v4i*)(unsigned long long)(const void*)g;
  as3_v4i* lp = (as3_v4i*)(unsigned)(unsigned long long)l;
  __builtin_amdgcn_global_load_async_to_lds_b128(gp, lp, 0, 0);
}
__device__ inline void async_drain() {
#if __has_builtin(__builtin_amdgcn_s_wait_asynccnt)
  __builtin_amdgcn_s_wait_asynccnt(0);
#else
  asm volatile("s_wait_asynccnt 0x0" ::: "memory");
#endif
}
#else
#define ASYNC_LDS 0
#endif

// Stage one 16x256 bf16 prior tile (8KB) into LDS; 256 threads x 32B each.
__device__ inline void stage_tile(__bf16* lbuf, const __bf16* gsrc, int tid) {
  const char* g = (const char*)gsrc + tid * 32;
  char* l = (char*)lbuf + tid * 32;
#if ASYNC_LDS
  async_cp16(g, l);
  async_cp16(g + 16, l + 16);
#else
  v4u a = *(const v4u*)g;
  v4u b = *(const v4u*)(g + 16);
  *(v4u*)l = a;
  *(v4u*)(l + 16) = b;
#endif
}

// ---------------- prep kernels ----------------

__global__ void k_f2bf(const float* __restrict__ src, __bf16* __restrict__ dst, int n) {
  int i = blockIdx.x * blockDim.x + threadIdx.x;
  if (i < n) dst[i] = (__bf16)src[i];
}

// wT[l][n][k] = bf16(W[l][k][n])  (B-matrix stored N-major, K-contiguous)
__global__ void k_wT(const float* __restrict__ W, __bf16* __restrict__ wT,
                     int Kd, int Nd, int layers) {
  int i = blockIdx.x * blockDim.x + threadIdx.x;
  int per = Kd * Nd;
  if (i >= layers * per) return;
  int l = i / per, r = i % per;
  int n = r / Kd, k = r % Kd;
  wT[i] = (__bf16)W[l * per + k * Nd + n];
}

__global__ void k_nb(const float* __restrict__ prior, float* __restrict__ nb) {
  int wave = (blockIdx.x * blockDim.x + threadIdx.x) >> 5;
  int lane = threadIdx.x & 31;
  if (wave >= K_) return;
  const float* row = prior + (size_t)wave * F_;
  float s = 0.f;
  for (int c = lane; c < F_; c += 32) { float v = row[c]; s += v * v; }
  #pragma unroll
  for (int m = 16; m; m >>= 1) s += __shfl_xor(s, m, 32);
  if (lane == 0) nb[wave] = s;
}

__global__ void k_init(float* __restrict__ jac, float* __restrict__ acc) {
  int i = blockIdx.x * blockDim.x + threadIdx.x;
  if (i < B_) jac[i] = 0.f;
  if (i < 2) acc[i] = 0.f;
}

// ---------------- coupling layer ----------------

#define FLOW_WAVES 2

__global__ void __launch_bounds__(FLOW_WAVES * 32)
k_flow(const float* __restrict__ xin, float* __restrict__ xout,
       __bf16* __restrict__ xbf,
       const __bf16* __restrict__ w1T,   // [H][HALF]
       const __bf16* __restrict__ wsT,   // [HALF][H]
       const __bf16* __restrict__ wtT,   // [HALF][H]
       const float* __restrict__ b1, const float* __restrict__ bs,
       const float* __restrict__ bt, float* __restrict__ jac, int is_last) {
  __shared__ __attribute__((aligned(16))) __bf16 hsh[FLOW_WAVES * 16 * H_];
  const int wave = threadIdx.x >> 5;
  const int lane = threadIdx.x & 31;
  const int colL = lane & 15;
  const int hi   = lane >> 4;
  const int rowbase = (blockIdx.x * FLOW_WAVES + wave) * 16;
  __bf16* hl = hsh + wave * 16 * H_;

  // A fragments of xa (16 rows x K=128): 4 fragments, held in VGPRs
  v16bf afr[4];
  {
    const __bf16* abase = xbf + (size_t)(rowbase + colL) * F_;
    #pragma unroll
    for (int kt = 0; kt < 4; ++kt) {
      int k0 = kt * 32 + hi * 8;
      afr[kt] = ldfrag(abase + k0, abase + k0 + 16);
    }
  }

  // h = relu(xa @ W1 + b1) -> LDS bf16 [16][512]
  for (int nt = 0; nt < H_ / 16; ++nt) {
    float bv = b1[nt * 16 + colL];
    v8f c;
    #pragma unroll
    for (int j = 0; j < 8; ++j) c[j] = bv;
    const __bf16* bbase = w1T + (size_t)(nt * 16 + colL) * HALF_;
    #pragma unroll
    for (int kt = 0; kt < 4; ++kt) {
      int k0 = kt * 32 + hi * 8;
      v16bf b = ldfrag(bbase + k0, bbase + k0 + 16);
      c = __builtin_amdgcn_wmma_f32_16x16x32_bf16(false, afr[kt], false, b,
                                                  (short)0, c, false, false);
    }
    int col = nt * 16 + colL;
    #pragma unroll
    for (int j = 0; j < 8; ++j) {
      float v = c[j] > 0.f ? c[j] : 0.f;
      hl[(j + 8 * hi) * H_ + col] = (__bf16)v;   // D-layout -> row-major LDS
    }
  }
  // LDS ops are in-order within a wave; hl is wave-private -> no barrier.

  float jacc[8];
  #pragma unroll
  for (int j = 0; j < 8; ++j) jacc[j] = 0.f;

  // s = h @ Ws + bs ; t = h @ Wt + bt ; elementwise epilogue
  for (int nt2 = 0; nt2 < HALF_ / 16; ++nt2) {
    int col = nt2 * 16 + colL;
    float bsv = bs[col], btv = bt[col];
    v8f cs, ct;
    #pragma unroll
    for (int j = 0; j < 8; ++j) { cs[j] = bsv; ct[j] = btv; }
    const __bf16* bsb = wsT + (size_t)col * H_;
    const __bf16* btb = wtT + (size_t)col * H_;
    const __bf16* arow = hl + colL * H_;
    for (int kt = 0; kt < H_ / 32; ++kt) {
      int k0 = kt * 32 + hi * 8;
      v16bf a  = ldfrag(arow + k0, arow + k0 + 16);
      v16bf bS = ldfrag(bsb + k0, bsb + k0 + 16);
      v16bf bT = ldfrag(btb + k0, btb + k0 + 16);
      cs = __builtin_amdgcn_wmma_f32_16x16x32_bf16(false, a, false, bS,
                                                   (short)0, cs, false, false);
      ct = __builtin_amdgcn_wmma_f32_16x16x32_bf16(false, a, false, bT,
                                                   (short)0, ct, false, false);
    }
    #pragma unroll
    for (int j = 0; j < 8; ++j) {
      int r = rowbase + j + 8 * hi;
      float s = cs[j];
      if (!is_last) s = tanhf(s);
      float t = ct[j];
      float xbv = xin[(size_t)r * F_ + HALF_ + col];
      float xav = xin[(size_t)r * F_ + col];
      float yb = xbv * __expf(s) + t;
      jacc[j] += s;
      xout[(size_t)r * F_ + col] = yb;            // new x = [yb | xa]
      xout[(size_t)r * F_ + HALF_ + col] = xav;
      xbf[(size_t)r * F_ + col] = (__bf16)yb;
      xbf[(size_t)r * F_ + HALF_ + col] = (__bf16)xav;
    }
  }

  // per-row jac: reduce across the 16 column lanes of each half
  #pragma unroll
  for (int j = 0; j < 8; ++j) {
    float s = jacc[j];
    s += __shfl_xor(s, 8, 16);
    s += __shfl_xor(s, 4, 16);
    s += __shfl_xor(s, 2, 16);
    s += __shfl_xor(s, 1, 16);
    if (colL == 0) jac[rowbase + j + 8 * hi] += s;  // unique row per lane
  }
}

// ---------------- VQ argmin (dominant GEMM, fused) ----------------
// Block = 8 waves sharing one K-slice and one code-tile stream (B reuse),
// each wave owns 32 rows (A pinned in 128 VGPRs). Prior tiles (8KB) are
// staged into LDS once per block, double-buffered, via async global->LDS.

#define VQ_WAVES 8
#define RPW 2   // 16-row subtiles per wave -> 32 rows

__global__ void __launch_bounds__(VQ_WAVES * 32)
k_vq(const __bf16* __restrict__ xbf, const __bf16* __restrict__ pbf,
     const float* __restrict__ nb, float* __restrict__ partb,
     int* __restrict__ parti) {
  __shared__ __attribute__((aligned(16))) __bf16 tile[2][16 * F_]; // 2 x 8KB
  const int tid  = threadIdx.x;
  const int wave = tid >> 5;
  const int lane = tid & 31;
  const int colL = lane & 15;
  const int hi   = lane >> 4;
  const int ROWGROUPS = B_ / (16 * RPW * VQ_WAVES);  // 32 (256 rows per block)
  const int slice    = blockIdx.x / ROWGROUPS;       // 0..KSLICES-1
  const int rowGroup = blockIdx.x % ROWGROUPS;
  const int rowbase  = (rowGroup * VQ_WAVES + wave) * (16 * RPW);

  v16bf afr[RPW][8];
  #pragma unroll
  for (int t = 0; t < RPW; ++t) {
    const __bf16* abase = xbf + (size_t)(rowbase + t * 16 + colL) * F_;
    #pragma unroll
    for (int kt = 0; kt < 8; ++kt) {
      int k0 = kt * 32 + hi * 8;
      afr[t][kt] = ldfrag(abase + k0, abase + k0 + 16);
    }
  }

  float best[RPW][8];
  int   bidx[RPW][8];
  #pragma unroll
  for (int t = 0; t < RPW; ++t)
    #pragma unroll
    for (int j = 0; j < 8; ++j) { best[t][j] = 3.4e38f; bidx[t][j] = 0; }

  const int nt0 = slice * NT_PER_SLICE;

  // prologue: stage first tile
  stage_tile(tile[0], pbf + (size_t)nt0 * 16 * F_, tid);
#if ASYNC_LDS
  async_drain();
#endif
  __syncthreads();

  for (int i = 0; i < NT_PER_SLICE; ++i) {
    const int nt = nt0 + i;
    const int cur = i & 1;
    // overlap: kick off async copy of next tile while computing this one
    if (i + 1 < NT_PER_SLICE)
      stage_tile(tile[cur ^ 1], pbf + (size_t)(nt + 1) * 16 * F_, tid);

    int code = nt * 16 + colL;
    float nbv = nb[code];
    const __bf16* bbase = tile[cur] + colL * F_;   // tile row = code-in-tile

    v8f acc[RPW];
    #pragma unroll
    for (int t = 0; t < RPW; ++t)
      #pragma unroll
      for (int j = 0; j < 8; ++j) acc[t][j] = 0.f;
    #pragma unroll
    for (int kt = 0; kt < 8; ++kt) {
      int k0 = kt * 32 + hi * 8;
      v16bf b = ldfrag(bbase + k0, bbase + k0 + 16);  // ds_load_b128 x2
      #pragma unroll
      for (int t = 0; t < RPW; ++t)
        acc[t] = __builtin_amdgcn_wmma_f32_16x16x32_bf16(false, afr[t][kt],
                      false, b, (short)0, acc[t], false, false);
    }
    #pragma unroll
    for (int t = 0; t < RPW; ++t)
      #pragma unroll
      for (int j = 0; j < 8; ++j) {
        float d = nbv - 2.0f * acc[t][j];        // na constant per row: drop
        bool lt = d < best[t][j];
        best[t][j] = lt ? d : best[t][j];
        bidx[t][j] = lt ? code : bidx[t][j];
      }

#if ASYNC_LDS
    async_drain();        // next buffer fully landed in LDS
#endif
    __syncthreads();      // all waves done reading cur; next buffer visible
  }

  // argmin across the 16 column lanes of each half
  #pragma unroll
  for (int t = 0; t < RPW; ++t)
    #pragma unroll
    for (int j = 0; j < 8; ++j) {
      float b = best[t][j]; int i = bidx[t][j];
      #pragma unroll
      for (int m = 8; m; m >>= 1) {
        float ob = __shfl_xor(b, m, 16);
        int   oi = __shfl_xor(i, m, 16);
        bool take = (ob < b) || (ob == b && oi < i);
        b = take ? ob : b;
        i = take ? oi : i;
      }
      if (colL == 0) {
        int r = rowbase + t * 16 + j + 8 * hi;
        partb[r * KSLICES + slice] = b;
        parti[r * KSLICES + slice] = i;
      }
    }
}

__global__ void k_amin(const float* __restrict__ partb,
                       const int* __restrict__ parti, int* __restrict__ amin) {
  int r = blockIdx.x * blockDim.x + threadIdx.x;
  if (r >= B_) return;
  float b = partb[r * KSLICES]; int i = parti[r * KSLICES];
  #pragma unroll
  for (int s = 1; s < KSLICES; ++s) {
    float ob = partb[r * KSLICES + s]; int oi = parti[r * KSLICES + s];
    bool take = (ob < b) || (ob == b && oi < i);
    b = take ? ob : b; i = take ? oi : i;
  }
  amin[r] = i;
}

// ---------------- loss ----------------

#define LOSS_RPW 8
__global__ void k_loss(const float* __restrict__ x, const float* __restrict__ prior,
                       const int* __restrict__ amin, const float* __restrict__ jac,
                       float* __restrict__ acc) {
  int wave = (blockIdx.x * blockDim.x + threadIdx.x) >> 5;
  int lane = threadIdx.x & 31;
  int r0 = wave * LOSS_RPW;
  float ssum = 0.f, jsum = 0.f;
  for (int rr = 0; rr < LOSS_RPW; ++rr) {
    int r = r0 + rr;
    const float* xr = x + (size_t)r * F_;
    const float* pr = prior + (size_t)amin[r] * F_;
    float s = 0.f;
    for (int c = lane; c < F_; c += 32) { float d = xr[c] - pr[c]; s += d * d; }
    #pragma unroll
    for (int m = 16; m; m >>= 1) s += __shfl_xor(s, m, 32);
    if (lane == 0) { ssum += s; jsum += jac[r]; }
  }
  if (lane == 0) {
    atomicAdd(&acc[0], ssum);
    atomicAdd(&acc[1], jsum);
  }
}

__global__ void k_fin(const float* __restrict__ acc, float* __restrict__ out_loss) {
  // loss_1 and loss_2 share the same forward value: 1.25 * mean(||x-t||^2/2) - mean(jac)
  *out_loss = 1.25f * acc[0] / (2.0f * (float)B_) - acc[1] / (float)B_;
}

// ---------------- launcher ----------------

extern "C" void kernel_launch(void* const* d_in, const int* in_sizes, int n_in,
                              void* d_out, int out_size, void* d_ws, size_t ws_size,
                              hipStream_t stream) {
  const float* inx   = (const float*)d_in[0];
  const float* W1    = (const float*)d_in[1];
  const float* b1    = (const float*)d_in[2];
  const float* Ws    = (const float*)d_in[3];
  const float* bs    = (const float*)d_in[4];
  const float* Wt    = (const float*)d_in[5];
  const float* bt    = (const float*)d_in[6];
  const float* prior = (const float*)d_in[7];

  char* ws = (char*)d_ws;
  size_t off = 0;
  auto take = [&](size_t bytes) { size_t o = off; off = (off + bytes + 255) & ~(size_t)255; return o; };
  float*  x1   = (float*)(ws + take((size_t)B_ * F_ * 4));
  __bf16* xbf  = (__bf16*)(ws + take((size_t)B_ * F_ * 2));
  __bf16* pbf  = (__bf16*)(ws + take((size_t)K_ * F_ * 2));
  __bf16* w1T  = (__bf16*)(ws + take((size_t)L_ * H_ * HALF_ * 2));
  __bf16* wsT  = (__bf16*)(ws + take((size_t)L_ * H_ * HALF_ * 2));
  __bf16* wtT  = (__bf16*)(ws + take((size_t)L_ * H_ * HALF_ * 2));
  float*  nb   = (float*)(ws + take((size_t)K_ * 4));
  float*  jac  = (float*)(ws + take((size_t)B_ * 4));
  float*  partb= (float*)(ws + take((size_t)B_ * KSLICES * 4));
  int*    parti= (int*)  (ws + take((size_t)B_ * KSLICES * 4));
  int*    amin = (int*)  (ws + take((size_t)B_ * 4));
  float*  acc  = (float*)(ws + take(8));

  float* outx = (float*)d_out;
  float* outloss = outx + (size_t)B_ * F_;

  // prep
  k_f2bf<<<(B_ * F_) / 256, 256, 0, stream>>>(inx, xbf, B_ * F_);
  k_f2bf<<<(K_ * F_) / 256, 256, 0, stream>>>(prior, pbf, K_ * F_);
  int wtot = L_ * H_ * HALF_;
  k_wT<<<(wtot + 255) / 256, 256, 0, stream>>>(W1, w1T, HALF_, H_, L_);
  k_wT<<<(wtot + 255) / 256, 256, 0, stream>>>(Ws, wsT, H_, HALF_, L_);
  k_wT<<<(wtot + 255) / 256, 256, 0, stream>>>(Wt, wtT, H_, HALF_, L_);
  k_nb<<<(K_ * 32) / 256, 256, 0, stream>>>(prior, nb);
  k_init<<<(B_ + 255) / 256, 256, 0, stream>>>(jac, acc);

  // flow layers (ping-pong f32 between ws and d_out's x region; last -> d_out)
  const float* cin = inx;
  for (int l = 0; l < L_; ++l) {
    float* cout = (l & 1) ? outx : x1;
    k_flow<<<B_ / (16 * FLOW_WAVES), FLOW_WAVES * 32, 0, stream>>>(
        cin, cout, xbf,
        w1T + (size_t)l * H_ * HALF_, wsT + (size_t)l * HALF_ * H_,
        wtT + (size_t)l * HALF_ * H_,
        b1 + (size_t)l * H_, bs + (size_t)l * HALF_, bt + (size_t)l * HALF_,
        jac, (l == L_ - 1) ? 1 : 0);
    cin = cout;
  }

  // VQ fused GEMM + argmin: 256 blocks = 32 row-groups x 8 K-slices
  k_vq<<<(B_ / (16 * RPW * VQ_WAVES)) * KSLICES, VQ_WAVES * 32, 0, stream>>>(
      xbf, pbf, nb, partb, parti);
  k_amin<<<(B_ + 255) / 256, 256, 0, stream>>>(partb, parti, amin);
  k_loss<<<(B_ / LOSS_RPW) * 32 / 256, 256, 0, stream>>>(outx, prior, amin, jac, acc);
  k_fin<<<1, 1, 0, stream>>>(acc, outloss);
}
